// GCN_18047452578705
// MI455X (gfx1250) — matrix operand order repaired and checked by reading, not compiled, not run on hardware
//
#include <hip/hip_runtime.h>
#include <hip/hip_bf16.h>

// ---------------------------------------------------------------------------
// CDNA5 (gfx1250) ChebConv GNN forward.
// Dense GEMMs: v_wmma_f32_16x16x32_bf16, bf16 operands pre-converted,
// A via global_load_b128, B staged in LDS (ds_load_b128), f32 accumulate.
// ---------------------------------------------------------------------------

typedef __attribute__((ext_vector_type(16))) __bf16 v16bf;
typedef __attribute__((ext_vector_type(8)))  __bf16 v8bf;
typedef __attribute__((ext_vector_type(8)))  float  v8f;

static __device__ __forceinline__ __bf16 f2bf(float f) {
    // round-to-nearest-even f32 -> bf16
    unsigned u = __builtin_bit_cast(unsigned, f);
    u += 0x7fffu + ((u >> 16) & 1u);
    unsigned short h = (unsigned short)(u >> 16);
    return __builtin_bit_cast(__bf16, h);
}

// ---------------------------- elementwise helpers ---------------------------

__global__ void k_zero(float* __restrict__ p, int n) {
    for (int i = blockIdx.x * blockDim.x + threadIdx.x; i < n; i += gridDim.x * blockDim.x)
        p[i] = 0.0f;
}

__global__ void k_neg_copy(float* __restrict__ out, const float* __restrict__ a, int n) {
    for (int i = blockIdx.x * blockDim.x + threadIdx.x; i < n; i += gridDim.x * blockDim.x)
        out[i] = -a[i];
}

__global__ void k_f32_to_bf16(const float* __restrict__ a, __bf16* __restrict__ o, int n) {
    for (int i = blockIdx.x * blockDim.x + threadIdx.x; i < n; i += gridDim.x * blockDim.x)
        o[i] = f2bf(a[i]);
}

__global__ void k_bias_relu(const float* __restrict__ acc, const float* __restrict__ b,
                            float* __restrict__ out, int n, int F) {
    for (int i = blockIdx.x * blockDim.x + threadIdx.x; i < n; i += gridDim.x * blockDim.x)
        out[i] = fmaxf(acc[i] + b[i % F], 0.0f);
}

// ------------------------------- graph prep --------------------------------

__global__ void k_scatter_deg(const int* __restrict__ src, const float* __restrict__ ea,
                              float* __restrict__ deg, int E) {
    int e = blockIdx.x * blockDim.x + threadIdx.x;
    if (e < E) atomicAdd(&deg[src[e]], ea[e]);
}

__global__ void k_dinv(const float* __restrict__ deg, float* __restrict__ dinv, int n) {
    int i = blockIdx.x * blockDim.x + threadIdx.x;
    if (i < n) {
        float d = deg[i];
        dinv[i] = (d > 0.0f) ? rsqrtf(d) : 0.0f;
    }
}

__global__ void k_edge_w(const int* __restrict__ src, const int* __restrict__ dst,
                         const float* __restrict__ ea, const float* __restrict__ dinv,
                         float* __restrict__ w, int E) {
    int e = blockIdx.x * blockDim.x + threadIdx.x;
    if (e < E) w[e] = -dinv[src[e]] * ea[e] * dinv[dst[e]];
}

// ----------------------- sparse propagation (scatter) ----------------------
// One wave per edge; lanes stripe the F features (coalesced). h rows are
// L2-resident (<=25.6MB vs 192MB L2); atomics resolve in L2.

__global__ void k_scatter_prop(const float* __restrict__ h, const int* __restrict__ src,
                               const int* __restrict__ dst, const float* __restrict__ w,
                               float* __restrict__ out, int E, int F, float scale) {
    int wave = (blockIdx.x * blockDim.x + threadIdx.x) >> 5;
    int lane = threadIdx.x & 31;
    if (wave >= E) return;
    int s = src[wave], d = dst[wave];
    float c = scale * w[wave];
    const float* __restrict__ hs = h + (size_t)s * F;
    float* __restrict__ od = out + (size_t)d * F;
    for (int f = lane; f < F; f += 32)
        atomicAdd(&od[f], c * hs[f]);
}

// ------------------------------ WMMA GEMM ----------------------------------
// C[N x Fout] += A16[N x FIN](bf16) * W16[FIN x Fout](bf16), f32 accumulate.
// Block = 8 waves sharing one 16-column tile of W staged in LDS (col-major).
// Per wave / K-step: 2x global_load_b128 (A) + 2x ds_load_b128 (B) + 1 WMMA.
// Fragment layouts per CDNA5 ISA 7.12.2; EXEC all-ones around every WMMA.

template <int FIN>
__global__ void __launch_bounds__(256)
k_wmma_gemm(const __bf16* __restrict__ A, const __bf16* __restrict__ W,
            float* __restrict__ C, int Fout, int Mtiles) {
    __shared__ __bf16 sW[16 * FIN];

    const int tid = threadIdx.x;
    const int tn  = blockIdx.y << 4;

    // Cooperative stage: sW[col][k] = W[k][tn+col]  (transposed -> col-major)
    for (int idx = tid; idx < 16 * FIN; idx += 256) {
        int col = idx / FIN;
        int k   = idx - col * FIN;
        sW[col * FIN + k] = W[(size_t)k * Fout + tn + col];
    }
    __syncthreads();

    const int wave  = tid >> 5;
    const int tileM = blockIdx.x * 8 + wave;
    if (tileM >= Mtiles) return;            // wave-uniform guard

    const int lane = tid & 31;
    const int l15  = lane & 15;
    const int half = lane >> 4;
    const int tm   = tileM << 4;

    float* Crow = C + (size_t)(tm + half * 8) * Fout + tn + l15;
    v8f c;
    #pragma unroll
    for (int r = 0; r < 8; ++r) c[r] = Crow[(size_t)r * Fout];

    const __bf16* __restrict__ Arow = A + (size_t)(tm + l15) * FIN;
    const __bf16*              Bcol = sW + l15 * FIN;

    #pragma unroll
    for (int kk = 0; kk < FIN; kk += 32) {
        v8bf alo = *reinterpret_cast<const v8bf*>(Arow + kk + half * 8);
        v8bf ahi = *reinterpret_cast<const v8bf*>(Arow + kk + 16 + half * 8);
        v8bf blo = *reinterpret_cast<const v8bf*>(Bcol + kk + half * 8);
        v8bf bhi = *reinterpret_cast<const v8bf*>(Bcol + kk + 16 + half * 8);
        v16bf a = __builtin_shufflevector(alo, ahi, 0,1,2,3,4,5,6,7,8,9,10,11,12,13,14,15);
        v16bf b = __builtin_shufflevector(blo, bhi, 0,1,2,3,4,5,6,7,8,9,10,11,12,13,14,15);
        c = __builtin_amdgcn_wmma_f32_16x16x32_bf16(
                /*neg_a=*/false, a, /*neg_b=*/false, b,
                /*c_mod=*/(short)0, c, /*reuse_a=*/false, /*reuse_b=*/false);
    }

    #pragma unroll
    for (int r = 0; r < 8; ++r) Crow[(size_t)r * Fout] = c[r];
}

// ------------------------------- pooling -----------------------------------

__global__ void k_pool_accum(const float* __restrict__ h, const int* __restrict__ batch,
                             float* __restrict__ pooled, int n, int F) {
    for (int i = blockIdx.x * blockDim.x + threadIdx.x; i < n; i += gridDim.x * blockDim.x) {
        int node = i / F, f = i % F;
        atomicAdd(&pooled[(size_t)batch[node] * F + f], h[i]);
    }
}

__global__ void k_pool_count(const int* __restrict__ batch, float* __restrict__ cnt, int N) {
    int i = blockIdx.x * blockDim.x + threadIdx.x;
    if (i < N) atomicAdd(&cnt[batch[i]], 1.0f);
}

__global__ void k_head(const float* __restrict__ pooled, const float* __restrict__ cnt,
                       const float* __restrict__ Wl, const float* __restrict__ bl,
                       float* __restrict__ out, int G, int F) {
    int g = blockIdx.x * blockDim.x + threadIdx.x;
    if (g >= G) return;
    float inv = 1.0f / fmaxf(cnt[g], 1.0f);
    float l0 = bl[0], l1 = bl[1];
    const float* __restrict__ p = pooled + (size_t)g * F;
    for (int j = 0; j < F; ++j) {
        float v = p[j] * inv;
        l0 += v * Wl[j * 2 + 0];
        l1 += v * Wl[j * 2 + 1];
    }
    float m   = fmaxf(l0, l1);
    float lse = m + logf(__expf(l0 - m) + __expf(l1 - m));
    out[g * 2 + 0] = l0 - lse;
    out[g * 2 + 1] = l1 - lse;
}

// ------------------------------- host side ---------------------------------

static inline int cdiv(int a, int b) { return (a + b - 1) / b; }

static void run_zero(float* p, int n, hipStream_t s) {
    k_zero<<<min(cdiv(n, 256), 4096), 256, 0, s>>>(p, n);
}

static void run_cvt(const float* a, __bf16* o, int n, hipStream_t s) {
    k_f32_to_bf16<<<min(cdiv(n, 256), 4096), 256, 0, s>>>(a, o, n);
}

static void launch_gemm(const __bf16* A16, const __bf16* W16, float* C,
                        int Fin, int Fout, int N, hipStream_t s) {
    const int Mtiles = N / 16;              // N % 16 == 0 (50000 = 16*3125)
    dim3 grid(cdiv(Mtiles, 8), Fout / 16);
    if (Fin == 128)
        k_wmma_gemm<128><<<grid, 256, 0, s>>>(A16, W16, C, Fout, Mtiles);
    else
        k_wmma_gemm<64><<<grid, 256, 0, s>>>(A16, W16, C, Fout, Mtiles);
}

// One ChebConv layer: Hout = relu( sum_k Tx_k @ Wk[k] + b )
static void cheb_layer(const float* Xin, int Fin, int Fout,
                       const float* Wk, const float* b,
                       const int* src, const int* dst, const float* w,
                       float* bufA, float* bufB, float* bufC, float* acc,
                       __bf16* A16, __bf16* W16,
                       float* Hout, int N, int E, hipStream_t stream) {
    const int scatter_blocks = cdiv(E * 32, 256);

    run_cvt(Wk, W16, 5 * Fin * Fout, stream);   // all K weight slices -> bf16
    run_zero(acc, N * Fout, stream);

    // k = 0: Tx0 = Xin
    run_cvt(Xin, A16, N * Fin, stream);
    launch_gemm(A16, W16, acc, Fin, Fout, N, stream);

    // k = 1: Tx1 = prop(Xin) -> bufA
    run_zero(bufA, N * Fin, stream);
    k_scatter_prop<<<scatter_blocks, 256, 0, stream>>>(Xin, src, dst, w, bufA, E, Fin, 1.0f);
    run_cvt(bufA, A16, N * Fin, stream);
    launch_gemm(A16, W16 + (size_t)1 * Fin * Fout, acc, Fin, Fout, N, stream);

    // k = 2..4: Tx_k = 2*prop(Tx_{k-1}) - Tx_{k-2}
    const float* t0 = Xin;
    float* t1 = bufA;
    float* rot[3] = {bufB, bufC, bufA};
    for (int k = 2; k < 5; ++k) {
        float* nx = rot[k - 2];
        k_neg_copy<<<min(cdiv(N * Fin, 256), 4096), 256, 0, stream>>>(nx, t0, N * Fin);
        k_scatter_prop<<<scatter_blocks, 256, 0, stream>>>(t1, src, dst, w, nx, E, Fin, 2.0f);
        run_cvt(nx, A16, N * Fin, stream);
        launch_gemm(A16, W16 + (size_t)k * Fin * Fout, acc, Fin, Fout, N, stream);
        t0 = t1;
        t1 = nx;
    }

    k_bias_relu<<<min(cdiv(N * Fout, 256), 4096), 256, 0, stream>>>(acc, b, Hout, N * Fout, Fout);
}

extern "C" void kernel_launch(void* const* d_in, const int* in_sizes, int n_in,
                              void* d_out, int out_size, void* d_ws, size_t ws_size,
                              hipStream_t stream) {
    const float* x     = (const float*)d_in[0];
    const int*   ei    = (const int*)d_in[1];
    const float* ea    = (const float*)d_in[2];
    const int*   batch = (const int*)d_in[3];
    const float* W1    = (const float*)d_in[4];
    const float* b1    = (const float*)d_in[5];
    const float* W2    = (const float*)d_in[6];
    const float* b2    = (const float*)d_in[7];
    const float* W3    = (const float*)d_in[8];
    const float* b3    = (const float*)d_in[9];
    const float* Wl    = (const float*)d_in[10];
    const float* bl    = (const float*)d_in[11];

    const int E  = in_sizes[2];              // 1,600,000
    const int N  = in_sizes[3];              // 50,000
    const int F  = in_sizes[0] / N;          // 128
    const int H1 = in_sizes[5];              // 64
    const int H2 = in_sizes[7];              // 64
    const int H3 = in_sizes[9];              // 128
    const int G  = out_size / 2;             // 500
    const int Gpad = (G + 3) & ~3;           // keep later buffers 16B-aligned

    const int* src = ei;
    const int* dst = ei + E;

    // Workspace layout (floats first, then bf16; all 16B-aligned)
    float* ws     = (float*)d_ws;
    float* deg    = ws;                 ws += N;
    float* dinv   = ws;                 ws += N;
    float* w      = ws;                 ws += E;
    float* bufA   = ws;                 ws += (size_t)N * F;
    float* bufB   = ws;                 ws += (size_t)N * F;
    float* bufC   = ws;                 ws += (size_t)N * F;
    float* acc    = ws;                 ws += (size_t)N * F;
    float* h1     = ws;                 ws += (size_t)N * H1;
    float* h2     = ws;                 ws += (size_t)N * H2;
    float* pooled = ws;                 ws += (size_t)G * H3;
    float* cnt    = ws;                 ws += Gpad;
    __bf16* A16   = (__bf16*)ws;        ws += (size_t)N * F / 2;   // N*F bf16
    __bf16* W16   = (__bf16*)ws;                                   // 5*128*128 bf16 max
    float* h3     = bufA;               // bufA dead after layer-3 GEMMs

    // ---- normalized Laplacian edge weights ----
    run_zero(deg, N, stream);
    k_scatter_deg<<<cdiv(E, 256), 256, 0, stream>>>(src, ea, deg, E);
    k_dinv<<<cdiv(N, 256), 256, 0, stream>>>(deg, dinv, N);
    k_edge_w<<<cdiv(E, 256), 256, 0, stream>>>(src, dst, ea, dinv, w, E);

    // ---- three ChebConv layers ----
    cheb_layer(x,  F,  H1, W1, b1, src, dst, w, bufA, bufB, bufC, acc, A16, W16, h1, N, E, stream);
    cheb_layer(h1, H1, H2, W2, b2, src, dst, w, bufA, bufB, bufC, acc, A16, W16, h2, N, E, stream);
    cheb_layer(h2, H2, H3, W3, b3, src, dst, w, bufA, bufB, bufC, acc, A16, W16, h3, N, E, stream);

    // ---- global mean pool + linear head + log_softmax ----
    run_zero(pooled, G * H3, stream);
    run_zero(cnt, G, stream);
    k_pool_accum<<<min(cdiv(N * H3, 256), 4096), 256, 0, stream>>>(h3, batch, pooled, N * H3, H3);
    k_pool_count<<<cdiv(N, 256), 256, 0, stream>>>(batch, cnt, N);
    k_head<<<cdiv(G, 128), 128, 0, stream>>>(pooled, cnt, Wl, bl, (float*)d_out, G, H3);
}
